// MultiHeadAttention_7138235646461
// MI455X (gfx1250) — matrix operand order repaired and checked by reading, not compiled, not run on hardware
//
#include <hip/hip_runtime.h>

// MI455X / gfx1250 — wave32, WMMA f32_16x16x32_f16.
// B=2, S=2048, E=1024, H=16, DH=64.

typedef __attribute__((ext_vector_type(16))) _Float16 v16h;
typedef __attribute__((ext_vector_type(8)))  _Float16 v8h;
typedef __attribute__((ext_vector_type(8)))  float    v8f;
typedef __attribute__((ext_vector_type(4)))  float    v4f;

#define B_   2
#define S_   2048
#define E_   1024
#define H_   16
#define DH_  64
#define NEG_INF_ (-1e9f)

__device__ __forceinline__ v8f wmma32(v16h a, v16h b, v8f c) {
  return __builtin_amdgcn_wmma_f32_16x16x32_f16(false, a, false, b, (short)0, c,
                                                false, false);
}

// ---------------- WMMA fragment helpers (ISA 7.12.2 wave32 layouts) ----------
// All loaders are per-lane CONTIGUOUS so they lower to b128 loads.

// A 16x32 f16, row-major source (element (m,k) at base[m*stride + k]).
// lane<16: M=lane, K{0..7,16..23}; lane>=16: M=lane-16, K{8..15,24..31}.
__device__ __forceinline__ v16h load_a_frag_f16(const _Float16* __restrict__ base,
                                                int row_stride) {
  const int lane = threadIdx.x & 31;
  const _Float16* r = base + (size_t)(lane & 15) * row_stride + ((lane >> 4) * 8);
  const v8h lo = *(const v8h*)r;
  const v8h hi = *(const v8h*)(r + 16);
  return __builtin_shufflevector(lo, hi, 0, 1, 2, 3, 4, 5, 6, 7,
                                 8, 9, 10, 11, 12, 13, 14, 15);
}

// Same A layout, fp32 source (softmax probabilities), convert on load.
__device__ __forceinline__ v16h load_a_frag_f32(const float* __restrict__ base,
                                                int row_stride) {
  const int lane = threadIdx.x & 31;
  const float* r = base + (size_t)(lane & 15) * row_stride + ((lane >> 4) * 8);
  const v4f x0 = *(const v4f*)r;
  const v4f x1 = *(const v4f*)(r + 4);
  const v4f x2 = *(const v4f*)(r + 16);
  const v4f x3 = *(const v4f*)(r + 20);
  v16h a;
#pragma unroll
  for (int i = 0; i < 4; ++i) {
    a[i]      = (_Float16)x0[i];
    a[4 + i]  = (_Float16)x1[i];
    a[8 + i]  = (_Float16)x2[i];
    a[12 + i] = (_Float16)x3[i];
  }
  return a;
}

// B 32x16 f16 from a K-CONTIGUOUS (N-major) source:
// element (k,n) at base[n*n_stride + k]. lane<16: N=lane, K{0..7,16..23}; lane>=16: N=lane-16, K{8..15,24..31}.
__device__ __forceinline__ v16h load_b_kcontig(const _Float16* __restrict__ base,
                                               int n_stride) {
  const int lane = threadIdx.x & 31;
  const _Float16* r = base + (size_t)(lane & 15) * n_stride + ((lane >> 4) * 8);
  const v8h lo = *(const v8h*)r;
  const v8h hi = *(const v8h*)(r + 16);
  return __builtin_shufflevector(lo, hi, 0, 1, 2, 3, 4, 5, 6, 7,
                                 8, 9, 10, 11, 12, 13, 14, 15);
}

// ---------------- Kernel 1: Y = relu(X @ W + b), f16, head-split ------------
// TRANSPOSE_STORE=false -> (B,H,S,DH); true -> (B,H,DH,S) (for V, so the
// attn@v B-operand is K-contiguous).
template <bool TRANSPOSE_STORE>
__global__ __launch_bounds__(256) void proj_gemm_relu(
    const float* __restrict__ X, const float* __restrict__ W,
    const float* __restrict__ bias, _Float16* __restrict__ dst) {
  __shared__ _Float16 As[64 * 32];    // (m,k) row-major, 4 KB
  __shared__ _Float16 Bt[128 * 32];   // (n,k) N-major (transposed), 8 KB

  const int m0   = blockIdx.x * 64;
  const int n0   = blockIdx.y * 128;
  const int tid  = threadIdx.x;
  const int wave = tid >> 5;
  const int ms   = wave & 3;   // 4 M strips of 16
  const int ns   = wave >> 2;  // 2 N strips of 64

  v8f acc[4] = {};

  for (int k0 = 0; k0 < E_; k0 += 32) {
    {  // stage A: vectorized f32 read, one v8h LDS store per thread
      const int r = tid >> 2, c = (tid & 3) * 8;
      const float* src = X + (size_t)(m0 + r) * E_ + k0 + c;
      __builtin_prefetch(src + 32, 0, 0);
      const v4f x0 = *(const v4f*)src;
      const v4f x1 = *(const v4f*)(src + 4);
      v8h hv;
#pragma unroll
      for (int i = 0; i < 4; ++i) {
        hv[i] = (_Float16)x0[i];
        hv[4 + i] = (_Float16)x1[i];
      }
      *(v8h*)&As[r * 32 + c] = hv;
    }
#pragma unroll
    for (int g = 0; g < 2; ++g) {  // stage B: coalesced read, transpose store
      const int idx = g * 2048 + tid * 8;
      const int kk = idx >> 7, nn = idx & 127;
      const float* src = W + (size_t)(k0 + kk) * E_ + n0 + nn;
      const v4f x0 = *(const v4f*)src;
      const v4f x1 = *(const v4f*)(src + 4);
#pragma unroll
      for (int i = 0; i < 4; ++i) {
        Bt[(nn + i) * 32 + kk]     = (_Float16)x0[i];
        Bt[(nn + 4 + i) * 32 + kk] = (_Float16)x1[i];
      }
    }
    __syncthreads();

    const v16h a = load_a_frag_f16(&As[(ms * 16) * 32], 32);
#pragma unroll
    for (int t = 0; t < 4; ++t) {
      const v16h b = load_b_kcontig(&Bt[(ns * 64 + t * 16) * 32], 32);
      acc[t] = wmma32(a, b, acc[t]);
    }
    __syncthreads();
  }

  // Epilogue: bias + relu, f16 store.
  const int lane = tid & 31;
  const int nloc = lane & 15;
  const int mb   = (lane >> 4) * 8;
#pragma unroll
  for (int t = 0; t < 4; ++t) {
    const int ng = n0 + ns * 64 + t * 16 + nloc;  // = h*64 + d
    const float bv = bias[ng];
    const int h = ng >> 6, d = ng & 63;
#pragma unroll
    for (int r = 0; r < 8; ++r) {
      const int mg = m0 + ms * 16 + mb + r;       // = b*S + s
      const int bb = mg >> 11, s = mg & 2047;
      const _Float16 hv = (_Float16)fmaxf(acc[t][r] + bv, 0.0f);
      if (TRANSPOSE_STORE)
        dst[((size_t)(bb * H_ + h) * DH_ + d) * S_ + s] = hv;
      else
        dst[((size_t)(bb * H_ + h) * S_ + s) * DH_ + d] = hv;
    }
  }
}

// ---------------- Kernel 2: fused scores -> mask -> softmax -> attn@v -------
// One block per (b*H+h, 16-query-row slab). Scores for 16 rows x 2048 keys
// live in 128 KB LDS (WGP has 320 KB); attn written to HBM exactly once.
__global__ __launch_bounds__(256) void attn_fused(
    const _Float16* __restrict__ qh, const _Float16* __restrict__ kh,
    const _Float16* __restrict__ vt, const float* __restrict__ mask,
    float* __restrict__ attn, float* __restrict__ outc) {
  __shared__ float P[16 * S_];            // 128 KB scores/probs
  __shared__ float red[256];              // softmax reductions
  __shared__ float partial[4 * 16 * 16];  // cross-wave K-split combine

  const int bh   = blockIdx.x;        // b*H + h
  const int m0   = blockIdx.y * 16;   // query row slab
  const int tid  = threadIdx.x;
  const int wave = tid >> 5;
  const int lane = tid & 31;
  const int nloc = lane & 15;
  const int mb   = (lane >> 4) * 8;
  const int bb   = bh >> 4;

  const _Float16* qbase = qh + ((size_t)bh * S_ + m0) * DH_;
  const _Float16* kbase = kh + (size_t)bh * S_ * DH_;        // (s,d): K-contig
  const _Float16* vbase = vt + (size_t)bh * DH_ * S_;        // (d,s): K-contig

  // ---- Phase 1: scores. Wave w covers key cols [w*256, w*256+256).
  const v16h aq0 = load_a_frag_f16(qbase, DH_);
  const v16h aq1 = load_a_frag_f16(qbase + 32, DH_);
  for (int t = 0; t < 16; ++t) {
    const int n0 = wave * 256 + t * 16;
    const _Float16* kp = kbase + (size_t)n0 * DH_;
    v8f c = {};
    c = wmma32(aq0, load_b_kcontig(kp, DH_), c);
    c = wmma32(aq1, load_b_kcontig(kp + 32, DH_), c);
    const int ng = n0 + nloc;
    const float mval = mask[bb * S_ + ng] * NEG_INF_;
#pragma unroll
    for (int r = 0; r < 8; ++r)
      P[(mb + r) * S_ + ng] = c[r] * 0.125f + mval;
  }
  __syncthreads();

  // ---- Phase 2: softmax (16 threads per row, 128 cols each).
  const int row = tid >> 4;
  const int seg = (tid & 15) * 128;
  float* pr = &P[row * S_ + seg];

  float lmax = -3.4e38f;
  for (int i = 0; i < 128; i += 4) {
    const v4f x = *(const v4f*)&pr[i];
    lmax = fmaxf(lmax, fmaxf(fmaxf(x[0], x[1]), fmaxf(x[2], x[3])));
  }
  red[tid] = lmax;
  __syncthreads();
  if ((tid & 15) == 0) {
    float m = red[tid];
#pragma unroll
    for (int i = 1; i < 16; ++i) m = fmaxf(m, red[tid + i]);
    red[tid] = m;
  }
  __syncthreads();
  const float rmax = red[row * 16];
  __syncthreads();

  float lsum = 0.0f;
  for (int i = 0; i < 128; ++i) {
    const float e = __expf(pr[i] - rmax);
    pr[i] = e;
    lsum += e;
  }
  red[tid] = lsum;
  __syncthreads();
  if ((tid & 15) == 0) {
    float s = 0.0f;
#pragma unroll
    for (int i = 0; i < 16; ++i) s += red[tid + i];
    red[tid] = s;
  }
  __syncthreads();
  const float inv = 1.0f / red[row * 16];

  // normalize in LDS and stream the fp32 probabilities to HBM (single write)
  float* arow = attn + ((size_t)bh * S_ + m0 + row) * S_ + seg;
  for (int i = 0; i < 128; i += 4) {
    v4f x = *(const v4f*)&pr[i];
    x *= inv;
    *(v4f*)&pr[i] = x;
    *(v4f*)&arow[i] = x;
  }
  __syncthreads();

  // ---- Phase 3: out(16x64) = P(16x2048) @ v(2048x64).
  // Wave w: N-tile (w&3), K-half (w>>2); combine pairs through LDS.
  const int t     = wave & 3;
  const int khalf = wave >> 2;
  const _Float16* vb = vbase + (size_t)(t * 16) * S_ + khalf * 1024;
  const float* pbase = &P[khalf * 1024];
  v8f acc = {};
  for (int kk = 0; kk < 1024; kk += 32) {
    const v16h a = load_a_frag_f32(pbase + kk, S_);
    const v16h b = load_b_kcontig(vb + kk, S_);
    acc = wmma32(a, b, acc);
  }
  if (khalf == 1) {
#pragma unroll
    for (int r = 0; r < 8; ++r)
      partial[t * 256 + (mb + r) * 16 + nloc] = acc[r];
  }
  __syncthreads();
  if (khalf == 0) {
    float* obase = outc + ((size_t)bh * S_ + m0) * DH_ + t * 16;  // flat (B,H,S,DH)
#pragma unroll
    for (int r = 0; r < 8; ++r)
      obase[(size_t)(mb + r) * DH_ + nloc] =
          acc[r] + partial[t * 256 + (mb + r) * 16 + nloc];
  }
}

// ---------------- Host side --------------------------------------------------
extern "C" void kernel_launch(void* const* d_in, const int* in_sizes, int n_in,
                              void* d_out, int out_size, void* d_ws, size_t ws_size,
                              hipStream_t stream) {
  const float* Q    = (const float*)d_in[0];
  const float* K    = (const float*)d_in[1];
  const float* V    = (const float*)d_in[2];
  const float* mask = (const float*)d_in[3];
  const float* Wq   = (const float*)d_in[4];
  const float* bq   = (const float*)d_in[5];
  const float* Wk   = (const float*)d_in[6];
  const float* bk   = (const float*)d_in[7];
  const float* Wv   = (const float*)d_in[8];
  const float* bv   = (const float*)d_in[9];

  float* out  = (float*)d_out;                  // concat: flat (B,H,S,DH) f32
  float* attn = out + (size_t)B_ * S_ * E_;     // attn:  (B,H,S,S) f32

  _Float16* qh = (_Float16*)d_ws;               // (B,H,S,DH) f16
  _Float16* kh = qh + (size_t)B_ * S_ * E_;     // (B,H,S,DH) f16
  _Float16* vt = kh + (size_t)B_ * S_ * E_;     // (B,H,DH,S) f16 (transposed)

  const dim3 blk(256);

  const dim3 g1(4096 / 64, E_ / 128);
  proj_gemm_relu<false><<<g1, blk, 0, stream>>>(Q, Wq, bq, qh);
  proj_gemm_relu<false><<<g1, blk, 0, stream>>>(K, Wk, bk, kh);
  proj_gemm_relu<true ><<<g1, blk, 0, stream>>>(V, Wv, bv, vt);

  const dim3 g2(B_ * H_, S_ / 16);
  attn_fused<<<g2, blk, 0, stream>>>(qh, kh, vt, mask, attn, out);
}
